// MultiHeadAttentionLayer_21990232555924
// MI455X (gfx1250) — matrix operand order repaired and checked by reading, not compiled, not run on hardware
//
#include <hip/hip_runtime.h>

// MHA: B=2, S=2048, D=1024, H=16, DK=64.  fp32 in/out, f16 WMMA compute.
// LDS-staged, double-buffered, 2x register-blocked WMMA pipeline (gfx1250 wave32).
// __launch_bounds__(256,1) lifts the VGPR cap so accumulators never spill.

#define Bb 2
#define Ss 2048
#define Dd 1024
#define Hh 16
#define DKk 64
#define Mrows (Bb * Ss)   // 4096
#define LSTR 72           // LDS row stride in halves: 144B = 16B-aligned, bank-conflict-free

typedef __attribute__((ext_vector_type(16))) _Float16 v16h;
typedef __attribute__((ext_vector_type(8)))  _Float16 v8h;
typedef __attribute__((ext_vector_type(4)))  _Float16 v4h;
typedef __attribute__((ext_vector_type(8)))  float    v8f;

__device__ __forceinline__ v16h cat8(v8h lo, v8h hi) {
    v16h r;
#pragma unroll
    for (int i = 0; i < 8; ++i) { r[i] = lo[i]; r[8 + i] = hi[i]; }
    return r;
}

__device__ __forceinline__ v8f wmma16(v16h a, v16h b, v8f c) {
    // D = A(16x32 f16) * B(32x16 f16) + C(16x16 f32)
    return __builtin_amdgcn_wmma_f32_16x16x32_f16(
        /*neg_a=*/false, a, /*neg_b=*/false, b,
        /*c_mod=*/(short)0, c, /*reuse_a=*/false, /*reuse_b=*/false);
}

// ---------------- elementwise convert (vectorized: float4 -> 4x f16) ----------------
__global__ void cvt_f32_f16_kernel(const float* __restrict__ src,
                                   _Float16* __restrict__ dst, int n4) {
    int i = blockIdx.x * blockDim.x + threadIdx.x;
    if (i < n4) {
        const float4 v = ((const float4*)src)[i];
        v4h o;
        o[0] = (_Float16)v.x; o[1] = (_Float16)v.y;
        o[2] = (_Float16)v.z; o[3] = (_Float16)v.w;
        ((v4h*)dst)[i] = o;
    }
}

// ---------------- tiled transpose convert: Wt[n][k] = (f16)W[k][n] ----------------
__global__ void transpose_cvt_kernel(const float* __restrict__ W,
                                     _Float16* __restrict__ Wt) {
    __shared__ _Float16 tile[32][33];
    const int x  = blockIdx.x * 32 + threadIdx.x;   // column of W (= n)
    const int y0 = blockIdx.y * 32;                 // row block of W (= k)
    for (int j = threadIdx.y; j < 32; j += 8)
        tile[j][threadIdx.x] = (_Float16)W[(size_t)(y0 + j) * Dd + x];
    __syncthreads();
    const int n0 = blockIdx.x * 32;
    for (int j = threadIdx.y; j < 32; j += 8)
        Wt[(size_t)(n0 + j) * Dd + y0 + threadIdx.x] = tile[threadIdx.x][j];
}

// ---------------- QKV projection GEMM (LDS double-buffered, 2x row-blocked) ----------------
// Block (8 waves) computes 256 rows x 64 cols; each wave owns two 16-row tiles
// (m0 and m0+128) sharing one LDS-staged 64x32 Wt tile -> 8 WMMAs per k-step
// for 4 B-tile LDS loads.
// mode 0: out[b][h][s][dk]   (q,k)     mode 1: out[b][h][dk][s]  (v, transposed)
__global__ void __launch_bounds__(256, 1)
proj_kernel(const _Float16* __restrict__ A,
            const _Float16* __restrict__ Wt,
            _Float16* __restrict__ out, int mode) {
    __shared__ alignas(16) _Float16 bs[2][64 * LSTR];
    const int tidb = threadIdx.x;
    const int w    = tidb >> 5;
    const int lane = tidb & 31;
    const int l16  = lane & 15;
    const int g    = lane >> 4;
    const int n0   = (blockIdx.x & 15) << 6;                 // 0..960
    const int m0   = (blockIdx.x >> 4) * 256 + w * 16;       // tile 0 rows
    const int m1   = m0 + 128;                               // tile 1 rows

    // cooperative weight-tile load: 64 cols x 32 halves = 256 chunks of 16B
    const int ccol = tidb >> 2;            // 0..63
    const int cchk = (tidb & 3) << 3;      // 0,8,16,24 halves
    const _Float16* wsrc = Wt + (size_t)(n0 + ccol) * Dd + cchk;

    const _Float16* arp0 = A + (size_t)(m0 + l16) * Dd;
    const _Float16* arp1 = A + (size_t)(m1 + l16) * Dd;

    auto coopload = [&](int buf, int kb) {
        *(v8h*)(&bs[buf][ccol * LSTR + cchk]) = *(const v8h*)(wsrc + kb);
    };
    auto loadA = [&](const _Float16* arp, int kb) -> v16h {
        return cat8(*(const v8h*)(arp + kb + g * 16),
                    *(const v8h*)(arp + kb + g * 16 + 8));
    };

    v8f acc0[4] = {v8f{}, v8f{}, v8f{}, v8f{}};
    v8f acc1[4] = {v8f{}, v8f{}, v8f{}, v8f{}};

    coopload(0, 0);
    v16h a0 = loadA(arp0, 0);
    v16h a1 = loadA(arp1, 0);
    __syncthreads();

    for (int it = 0; it < Dd / 32; ++it) {
        const int kb = it * 32;
        v16h a0n = a0, a1n = a1;
        if (it + 1 < Dd / 32) {
            coopload((it + 1) & 1, kb + 32);
            a0n = loadA(arp0, kb + 32);
            a1n = loadA(arp1, kb + 32);
            __builtin_prefetch(arp0 + kb + 128, 0, 3);
            __builtin_prefetch(arp1 + kb + 128, 0, 3);
        }
        const _Float16* bb = &bs[it & 1][0];
#pragma unroll
        for (int t = 0; t < 4; ++t) {
            const _Float16* bp = bb + (t * 16 + l16) * LSTR + g * 16;
            v16h b = cat8(*(const v8h*)(bp), *(const v8h*)(bp + 8));
            acc0[t] = wmma16(a0, b, acc0[t]);
            acc1[t] = wmma16(a1, b, acc1[t]);
        }
        __syncthreads();
        a0 = a0n; a1 = a1n;
    }

#pragma unroll
    for (int half = 0; half < 2; ++half) {
        const v8f* acc = half ? acc1 : acc0;
        const int mb   = half ? m1 : m0;
#pragma unroll
        for (int t = 0; t < 4; ++t) {
#pragma unroll
            for (int r = 0; r < 8; ++r) {
                int m = mb + r + 8 * g;           // global row (b*S+s)
                int n = n0 + t * 16 + l16;        // global col (h*DK+dk)
                int b = m >> 11, s = m & (Ss - 1);
                int hh = n >> 6, dk = n & (DKk - 1);
                _Float16 val = (_Float16)acc[t][r];
                if (mode == 0)
                    out[(((b * Hh + hh) * Ss) + s) * DKk + dk] = val;
                else
                    out[(((b * Hh + hh) * DKk) + dk) * Ss + s] = val;
            }
        }
    }
}

// ---------------- causal flash attention (LDS-staged K/V, 2 query tiles / wave) --------------
// Block = 256 queries of one (b,h): wave w owns query tiles qtA = qb*16+w and
// qtB = qtA+8.  Per 32-key chunk the K tile (32x64) and V tile (64x32,
// transposed) are staged once in LDS; K A-operand and V B-operand registers
// feed both query tiles' WMMAs.  Scores are computed transposed
// (S^T = K @ Q^T): softmax is per-lane and P lands directly in the A-operand
// layout of P@V.
__global__ void __launch_bounds__(256, 1)
attn_kernel(const _Float16* __restrict__ qf,
            const _Float16* __restrict__ kf,
            const _Float16* __restrict__ vt,
            _Float16* __restrict__ ctx) {
    __shared__ alignas(16) _Float16 sm[2][(32 + 64) * LSTR];  // K tile | V tile
    const int tidb = threadIdx.x;
    const int w    = tidb >> 5;
    const int lane = tidb & 31;
    const int l16  = lane & 15;
    const int g    = lane >> 4;

    const int qblock = blockIdx.x & 7;                // 256-query block (8 per b,h)
    const int h      = (blockIdx.x >> 3) & (Hh - 1);
    const int b      = blockIdx.x >> 7;

    const _Float16* qp = qf + (size_t)(b * Hh + h) * Ss * DKk;  // [s][dk]
    const _Float16* kp = kf + (size_t)(b * Hh + h) * Ss * DKk;  // [t][dk]
    const _Float16* vp = vt + (size_t)(b * Hh + h) * DKk * Ss;  // [dk][s]

    const int qtA = qblock * 16 + w;
    const int qtB = qtA + 8;
    const int qgA = qtA * 16 + l16;
    const int qgB = qtB * 16 + l16;

    // resident Q^T B-operand registers for both tiles
    const _Float16* qrA = qp + (size_t)qgA * DKk;
    const _Float16* qrB = qp + (size_t)qgB * DKk;
    v16h qbA0 = cat8(*(const v8h*)(qrA + g * 16),      *(const v8h*)(qrA + g * 16 + 8));
    v16h qbA1 = cat8(*(const v8h*)(qrA + 32 + g * 16), *(const v8h*)(qrA + 32 + g * 16 + 8));
    v16h qbB0 = cat8(*(const v8h*)(qrB + g * 16),      *(const v8h*)(qrB + g * 16 + 8));
    v16h qbB1 = cat8(*(const v8h*)(qrB + 32 + g * 16), *(const v8h*)(qrB + 32 + g * 16 + 8));

    // cooperative K/V chunk load: K 32x64 halves (256x16B), V 64x32 halves (256x16B)
    const int krow_c = tidb >> 3;          // 0..31 key row
    const int kpart  = (tidb & 7) << 3;    // 0..56 halves
    const int vrow_c = tidb >> 2;          // 0..63 dk row
    const int vchk   = (tidb & 3) << 3;    // 0..24 halves

    auto coopload = [&](int buf, int ks) {
        *(v8h*)(&sm[buf][krow_c * LSTR + kpart]) =
            *(const v8h*)(kp + (size_t)(ks + krow_c) * DKk + kpart);
        *(v8h*)(&sm[buf][(32 + vrow_c) * LSTR + vchk]) =
            *(const v8h*)(vp + (size_t)vrow_c * Ss + ks + vchk);
    };

    v8f oA[4] = {v8f{}, v8f{}, v8f{}, v8f{}};
    v8f oB[4] = {v8f{}, v8f{}, v8f{}, v8f{}};
    float mrowA = -1e30f, lrowA = 0.0f;
    float mrowB = -1e30f, lrowB = 0.0f;
    const float scale = 0.125f;            // 1/sqrt(64)

    const int nchA    = (qtA >> 1) + 1;    // causal chunk counts
    const int nchB    = (qtB >> 1) + 1;    // nchA <= nchB
    const int nch_blk = 8 * qblock + 8;    // chunks the block stages

    // online-softmax update for one 32-key chunk of one query tile
    auto softmax_update = [&](const v8f& st0, const v8f& st1, int ks, int qg,
                              float& mrow, float& lrow, v16h& pa) -> float {
        float x0[8], x1[8];
        float cm = -1e30f;
#pragma unroll
        for (int r = 0; r < 8; ++r) {
            int key0 = ks + r + 8 * g;
            int key1 = ks + 16 + r + 8 * g;
            x0[r] = (key0 <= qg) ? st0[r] * scale : -1e30f;
            x1[r] = (key1 <= qg) ? st1[r] * scale : -1e30f;
            cm = fmaxf(cm, fmaxf(x0[r], x1[r]));
        }
        cm = fmaxf(cm, __shfl_xor(cm, 16, 32));
        float mnew = fmaxf(mrow, cm);
        float f = __expf(mrow - mnew);
        float csum = 0.0f;
#pragma unroll
        for (int r = 0; r < 8; ++r) {
            float p0 = __expf(x0[r] - mnew);
            float p1 = __expf(x1[r] - mnew);
            csum += p0 + p1;
            pa[r]     = (_Float16)p0;
            pa[8 + r] = (_Float16)p1;
        }
        csum += __shfl_xor(csum, 16, 32);
        lrow = lrow * f + csum;
        mrow = mnew;
        return f;
    };

    coopload(0, 0);
    __syncthreads();

    for (int c = 0; c < nch_blk; ++c) {
        if (c + 1 < nch_blk) {
            coopload((c + 1) & 1, (c + 1) * 32);
            if (c + 2 < nch_blk)
                __builtin_prefetch(kp + (size_t)((c + 2) * 32 + krow_c) * DKk + kpart, 0, 3);
        }
        if (c < nchB) {                    // superset guard (wave-uniform)
            const _Float16* kt    = &sm[c & 1][0];
            const _Float16* vtile = &sm[c & 1][32 * LSTR];
            const int ks = c * 32;
            const bool activeA = (c < nchA);

            // S^T tiles: K A-operand registers shared by both query tiles
            v8f stA[2], stB[2];
#pragma unroll
            for (int j = 0; j < 2; ++j) {
                const _Float16* kr = kt + (j * 16 + l16) * LSTR;
                v16h ka0 = cat8(*(const v8h*)(kr + g * 16),      *(const v8h*)(kr + g * 16 + 8));
                v16h ka1 = cat8(*(const v8h*)(kr + 32 + g * 16), *(const v8h*)(kr + 32 + g * 16 + 8));
                if (activeA) {
                    v8f cc = {};
                    cc = wmma16(ka0, qbA0, cc);
                    cc = wmma16(ka1, qbA1, cc);
                    stA[j] = cc;
                }
                v8f cc = {};
                cc = wmma16(ka0, qbB0, cc);
                cc = wmma16(ka1, qbB1, cc);
                stB[j] = cc;
            }

            v16h paA, paB;
            float fA = 1.0f;
            if (activeA)
                fA = softmax_update(stA[0], stA[1], ks, qgA, mrowA, lrowA, paA);
            float fB = softmax_update(stB[0], stB[1], ks, qgB, mrowB, lrowB, paB);

            // rescale running outputs (per-query factor via lane broadcast)
            if (activeA) {
#pragma unroll
                for (int r = 0; r < 8; ++r) {
                    float fr = __shfl(fA, r + 8 * g, 32);
                    oA[0][r] *= fr; oA[1][r] *= fr; oA[2][r] *= fr; oA[3][r] *= fr;
                }
            }
#pragma unroll
            for (int r = 0; r < 8; ++r) {
                float fr = __shfl(fB, r + 8 * g, 32);
                oB[0][r] *= fr; oB[1][r] *= fr; oB[2][r] *= fr; oB[3][r] *= fr;
            }

            // out += P(16x32) @ V(32x64): V B-operand registers shared by both tiles
#pragma unroll
            for (int t = 0; t < 4; ++t) {
                const _Float16* vr = vtile + (t * 16 + l16) * LSTR + g * 8;
                v16h vb = cat8(*(const v8h*)(vr), *(const v8h*)(vr + 16));
                if (activeA) oA[t] = wmma16(paA, vb, oA[t]);
                oB[t] = wmma16(paB, vb, oB[t]);
            }
        }
        __syncthreads();
    }

    // normalize and store context f16 [b][s][h*64+dk]
    auto store_ctx = [&](int qt, const v8f* o, float lrow) {
#pragma unroll
        for (int r = 0; r < 8; ++r) {
            float li  = __shfl(lrow, r + 8 * g, 32);
            float inv = 1.0f / li;
            int s = qt * 16 + r + 8 * g;
#pragma unroll
            for (int t = 0; t < 4; ++t) {
                int d = h * DKk + t * 16 + l16;
                ctx[((size_t)(b * Ss + s)) * Dd + d] = (_Float16)(o[t][r] * inv);
            }
        }
    };
    store_ctx(qtA, oA, lrowA);
    store_ctx(qtB, oB, lrowB);
}

// ---------------- output projection + bias (fp32 out, 2x row-blocked) ----------------
__global__ void __launch_bounds__(256, 1)
outproj_kernel(const _Float16* __restrict__ A,
               const _Float16* __restrict__ Wt,
               const float* __restrict__ bias,
               float* __restrict__ out) {
    __shared__ alignas(16) _Float16 bs[2][64 * LSTR];
    const int tidb = threadIdx.x;
    const int w    = tidb >> 5;
    const int lane = tidb & 31;
    const int l16  = lane & 15;
    const int g    = lane >> 4;
    const int n0   = (blockIdx.x & 15) << 6;
    const int m0   = (blockIdx.x >> 4) * 256 + w * 16;
    const int m1   = m0 + 128;

    const int ccol = tidb >> 2;
    const int cchk = (tidb & 3) << 3;
    const _Float16* wsrc = Wt + (size_t)(n0 + ccol) * Dd + cchk;
    const _Float16* arp0 = A + (size_t)(m0 + l16) * Dd;
    const _Float16* arp1 = A + (size_t)(m1 + l16) * Dd;

    auto coopload = [&](int buf, int kb) {
        *(v8h*)(&bs[buf][ccol * LSTR + cchk]) = *(const v8h*)(wsrc + kb);
    };
    auto loadA = [&](const _Float16* arp, int kb) -> v16h {
        return cat8(*(const v8h*)(arp + kb + g * 16),
                    *(const v8h*)(arp + kb + g * 16 + 8));
    };

    v8f acc0[4] = {v8f{}, v8f{}, v8f{}, v8f{}};
    v8f acc1[4] = {v8f{}, v8f{}, v8f{}, v8f{}};

    coopload(0, 0);
    v16h a0 = loadA(arp0, 0);
    v16h a1 = loadA(arp1, 0);
    __syncthreads();

    for (int it = 0; it < Dd / 32; ++it) {
        const int kb = it * 32;
        v16h a0n = a0, a1n = a1;
        if (it + 1 < Dd / 32) {
            coopload((it + 1) & 1, kb + 32);
            a0n = loadA(arp0, kb + 32);
            a1n = loadA(arp1, kb + 32);
            __builtin_prefetch(arp0 + kb + 128, 0, 3);
            __builtin_prefetch(arp1 + kb + 128, 0, 3);
        }
        const _Float16* bb = &bs[it & 1][0];
#pragma unroll
        for (int t = 0; t < 4; ++t) {
            const _Float16* bp = bb + (t * 16 + l16) * LSTR + g * 16;
            v16h b = cat8(*(const v8h*)(bp), *(const v8h*)(bp + 8));
            acc0[t] = wmma16(a0, b, acc0[t]);
            acc1[t] = wmma16(a1, b, acc1[t]);
        }
        __syncthreads();
        a0 = a0n; a1 = a1n;
    }

#pragma unroll
    for (int half = 0; half < 2; ++half) {
        const v8f* acc = half ? acc1 : acc0;
        const int mb   = half ? m1 : m0;
#pragma unroll
        for (int t = 0; t < 4; ++t) {
            int n = n0 + t * 16 + l16;
            float bv = bias[n];
#pragma unroll
            for (int r = 0; r < 8; ++r) {
                int m = mb + r + 8 * g;
                out[(size_t)m * Dd + n] = acc[t][r] + bv;
            }
        }
    }
}

// ---------------- launcher ----------------
extern "C" void kernel_launch(void* const* d_in, const int* in_sizes, int n_in,
                              void* d_out, int out_size, void* d_ws, size_t ws_size,
                              hipStream_t stream) {
    (void)in_sizes; (void)n_in; (void)out_size; (void)ws_size;
    const float* Q  = (const float*)d_in[0];
    const float* K  = (const float*)d_in[1];
    const float* V  = (const float*)d_in[2];
    const float* Wq = (const float*)d_in[3];
    const float* Wk = (const float*)d_in[4];
    const float* Wv = (const float*)d_in[5];
    const float* Wo = (const float*)d_in[6];
    const float* bo = (const float*)d_in[7];
    float* out = (float*)d_out;

    char* ws = (char*)d_ws;
    const size_t MB = 1u << 20;
    _Float16* Qh  = (_Float16*)(ws + 0 * MB);   // 8 MB  (reused as ctx later)
    _Float16* Kh  = (_Float16*)(ws + 8 * MB);   // 8 MB
    _Float16* Vh  = (_Float16*)(ws + 16 * MB);  // 8 MB
    _Float16* Wqt = (_Float16*)(ws + 24 * MB);  // 2 MB
    _Float16* Wkt = (_Float16*)(ws + 26 * MB);  // 2 MB
    _Float16* Wvt = (_Float16*)(ws + 28 * MB);  // 2 MB
    _Float16* Wot = (_Float16*)(ws + 30 * MB);  // 2 MB
    _Float16* qf  = (_Float16*)(ws + 32 * MB);  // 8 MB  [B,H,S,DK]
    _Float16* kf  = (_Float16*)(ws + 40 * MB);  // 8 MB  [B,H,S,DK]
    _Float16* vtw = (_Float16*)(ws + 48 * MB);  // 8 MB  [B,H,DK,S]
    _Float16* ctx = Qh;                          // reuse: Qh dead after q-proj

    const int n4 = (Mrows * Dd) / 4;             // 1,048,576 float4 groups
    const int cvtGrid = n4 / 256;
    hipLaunchKernelGGL(cvt_f32_f16_kernel, dim3(cvtGrid), dim3(256), 0, stream, Q, Qh, n4);
    hipLaunchKernelGGL(cvt_f32_f16_kernel, dim3(cvtGrid), dim3(256), 0, stream, K, Kh, n4);
    hipLaunchKernelGGL(cvt_f32_f16_kernel, dim3(cvtGrid), dim3(256), 0, stream, V, Vh, n4);

    dim3 tg(Dd / 32, Dd / 32), tb(32, 8);
    hipLaunchKernelGGL(transpose_cvt_kernel, tg, tb, 0, stream, Wq, Wqt);
    hipLaunchKernelGGL(transpose_cvt_kernel, tg, tb, 0, stream, Wk, Wkt);
    hipLaunchKernelGGL(transpose_cvt_kernel, tg, tb, 0, stream, Wv, Wvt);
    hipLaunchKernelGGL(transpose_cvt_kernel, tg, tb, 0, stream, Wo, Wot);

    // GEMMs: 256 blocks x 256 threads (8 waves/block, 2 row tiles each)
    hipLaunchKernelGGL(proj_kernel, dim3(256), dim3(256), 0, stream, Qh, Wqt, qf, 0);
    hipLaunchKernelGGL(proj_kernel, dim3(256), dim3(256), 0, stream, Kh, Wkt, kf, 0);
    hipLaunchKernelGGL(proj_kernel, dim3(256), dim3(256), 0, stream, Vh, Wvt, vtw, 1);

    // attention: 256 blocks (2 x 16 x 8), 2 query tiles per wave
    hipLaunchKernelGGL(attn_kernel, dim3(256), dim3(256), 0, stream, qf, kf, vtw, ctx);

    hipLaunchKernelGGL(outproj_kernel, dim3(256), dim3(256), 0, stream, ctx, Wot, bo, out);
}
